// FractalFieldClassifier_19645180412080
// MI455X (gfx1250) — matrix operand order repaired and checked by reading, not compiled, not run on hardware
//
#include <hip/hip_runtime.h>
#include <math.h>

// ---------------- types for CDNA5 WMMA ----------------
typedef __bf16 bf16_t;
typedef __attribute__((ext_vector_type(16))) __bf16 v16bf;
typedef __attribute__((ext_vector_type(8)))  __bf16 v8bf;
typedef __attribute__((ext_vector_type(2)))  __bf16 v2bf;
typedef __attribute__((ext_vector_type(8)))  float  v8f;

union Frag16 {
    v16bf v;
    v8bf  h[2];
};

__device__ __forceinline__ unsigned short f2bf_bits(float f) {
    unsigned u = __builtin_bit_cast(unsigned, f);
    unsigned r = u + 0x7fffu + ((u >> 16) & 1u);   // round-to-nearest-even
    return (unsigned short)(r >> 16);
}
__device__ __forceinline__ bf16_t f2bf(float f) {
    return __builtin_bit_cast(bf16_t, f2bf_bits(f));
}
// pack two f32 -> two bf16 in one u32 (use HW packed cvt when available)
__device__ __forceinline__ unsigned pack_bf16x2(float lo, float hi) {
#if __has_builtin(__builtin_amdgcn_cvt_pk_bf16_f32)
    v2bf p = __builtin_amdgcn_cvt_pk_bf16_f32(lo, hi);
    return __builtin_bit_cast(unsigned, p);
#else
    return ((unsigned)f2bf_bits(hi) << 16) | f2bf_bits(lo);
#endif
}

#define BATCH 32
#define CCH   64
#define HWD   64
#define NSTEP 30
#define NCLS  10
#define EPSV  1e-6f
#define LPAD  72   // padded ci-stride in LDS (bank spreading, keeps 16B align)

// =====================================================================
// 1) Encoder: zr0 = conv_wrap(x, enc_w) + enc_b ; zi0 = 0 ; injr = zr0
// =====================================================================
__global__ void __launch_bounds__(256)
encode_kernel(const float* __restrict__ x, const float* __restrict__ ew,
              const float* __restrict__ eb, float* __restrict__ injr,
              float* __restrict__ zr0, float* __restrict__ zi0)
{
    const int tid = threadIdx.x;
    const int y = blockIdx.x, b = blockIdx.y;
    const int xx = tid & 63;
    const int cgrp = tid >> 6;           // 0..3 -> 16 channels each

    float p[9];
#pragma unroll
    for (int tap = 0; tap < 9; ++tap) {
        int yy = (y + tap / 3 - 1) & 63;
        int xw = (xx + tap % 3 - 1 + 64) & 63;
        p[tap] = x[((size_t)b * HWD + yy) * HWD + xw];
    }
#pragma unroll
    for (int k = 0; k < 16; ++k) {
        int c = cgrp * 16 + k;
        float acc = eb[c];
#pragma unroll
        for (int tap = 0; tap < 9; ++tap) acc += p[tap] * ew[c * 9 + tap];
        size_t g = (((size_t)b * CCH + c) * HWD + y) * HWD + xx;
        injr[g] = acc; zr0[g] = acc; zi0[g] = 0.f;
    }
}

// =====================================================================
// 2) Spectral norm of Wc = wr + i*wi (64 x 576) via power iteration
// =====================================================================
__global__ void __launch_bounds__(576)
spectral_kernel(const float* __restrict__ wr, const float* __restrict__ wi,
                float* __restrict__ sigma)
{
    __shared__ float ur[64], ui[64];
    __shared__ float vr[576], vi[576];
    __shared__ float red[1024];
    const int tid = threadIdx.x;

    vr[tid] = 1.f; vi[tid] = 0.f;
    __syncthreads();

    for (int it = 0; it < 64; ++it) {
        if (tid < 64) {                       // u = W v
            float ar = 0.f, ai = 0.f;
            for (int k = 0; k < 576; ++k) {
                float a = wr[tid * 576 + k], c = wi[tid * 576 + k];
                ar += a * vr[k] - c * vi[k];
                ai += a * vi[k] + c * vr[k];
            }
            ur[tid] = ar; ui[tid] = ai;
        }
        __syncthreads();
        float nr = 0.f, ni = 0.f;             // v = W^H u
        for (int co = 0; co < 64; ++co) {
            float a = wr[co * 576 + tid], c = wi[co * 576 + tid];
            nr += a * ur[co] + c * ui[co];
            ni += a * ui[co] - c * ur[co];
        }
        __syncthreads();
        red[tid] = nr * nr + ni * ni;
        if (tid < 448) red[576 + tid] = 0.f;
        __syncthreads();
        for (int s = 512; s >= 1; s >>= 1) {
            if (tid < s) red[tid] += red[tid + s];
            __syncthreads();
        }
        float inv = rsqrtf(red[0] + 1e-30f);
        vr[tid] = nr * inv; vi[tid] = ni * inv;
        __syncthreads();
    }
    if (tid < 64) {                           // sigma = ||W v||
        float ar = 0.f, ai = 0.f;
        for (int k = 0; k < 576; ++k) {
            float a = wr[tid * 576 + k], c = wi[tid * 576 + k];
            ar += a * vr[k] - c * vi[k];
            ai += a * vi[k] + c * vr[k];
        }
        red[tid] = ar * ar + ai * ai;
    }
    __syncthreads();
    if (tid == 0) {
        float s = 0.f;
        for (int i = 0; i < 64; ++i) s += red[i];
        sigma[0] = sqrtf(s);
    }
}

// =====================================================================
// 3) Pack normalized weights to bf16 in WMMA K-order [co][tap][ci];
//    also zero per-step reduction slots.
// =====================================================================
__global__ void __launch_bounds__(256)
prep_kernel(const float* __restrict__ wr, const float* __restrict__ wi,
            const float* __restrict__ sigma, bf16_t* __restrict__ wrn,
            bf16_t* __restrict__ win, bf16_t* __restrict__ wineg,
            float* __restrict__ dz2, float* __restrict__ nrm2)
{
    const int idx = blockIdx.x * 256 + threadIdx.x;      // 36864 total
    const float inv = 1.f / sigma[0];
    if (idx < CCH * 9 * CCH) {
        int ci  = idx & 63;
        int tap = (idx >> 6) % 9;
        int co  = idx / (9 * 64);
        float a = wr[((size_t)co * CCH + ci) * 9 + tap] * inv;
        float c = wi[((size_t)co * CCH + ci) * 9 + tap] * inv;
        wrn[idx]   = f2bf(a);
        win[idx]   = f2bf(c);
        wineg[idx] = f2bf(-c);
    }
    if (idx < NSTEP) { dz2[idx] = 0.f; nrm2[idx] = 0.f; }
}

// =====================================================================
// 4) One iteration step: complex conv via bf16 WMMA implicit GEMM +
//    magnitude nonlinearity + relaxation + global norm partials.
//    Grid: (H=64, B=32) ; block 256 (8 waves); one row per workgroup.
// =====================================================================
__global__ void __launch_bounds__(256)
step_kernel(const float* __restrict__ zin_r, const float* __restrict__ zin_i,
            float* __restrict__ zout_r, float* __restrict__ zout_i,
            const float* __restrict__ injr,
            const bf16_t* __restrict__ wrn, const bf16_t* __restrict__ win,
            const bf16_t* __restrict__ wineg,
            const float* __restrict__ mod_b, const float* __restrict__ alpha_p,
            float* __restrict__ dz2_slot, float* __restrict__ nrm2_slot)
{
    __shared__ bf16_t sZr[3][HWD][LPAD];
    __shared__ bf16_t sZi[3][HWD][LPAD];

    const int tid = threadIdx.x;
    const int y = blockIdx.x;
    const int b = blockIdx.y;
    const float alpha = alpha_p[0];

    // WGP-scope prefetch of weight matrices (L2 resident, warm L0)
    for (int off = tid; off < 1152; off += 256) {
        __builtin_prefetch((const char*)wrn   + off * 64, 0, 0);
        __builtin_prefetch((const char*)win   + off * 64, 0, 0);
        __builtin_prefetch((const char*)wineg + off * 64, 0, 0);
    }

    // Stage 3 wrap rows (zr, zi) as bf16 into LDS, layout [row][x][ci].
    // ci-fast over lanes -> conflict-free DS stores (packed b32 pairs).
    for (int e = tid; e < 3 * HWD * 32; e += 256) {
        int cp = e & 31;                // ci pair
        int xx = (e >> 5) & 63;
        int r  = e >> 11;               // 0..2
        int yy = (y + r - 1) & 63;
        int ci = cp * 2;
        size_t g0 = (((size_t)b * CCH + ci) * HWD + yy) * HWD + xx;
        float a0 = zin_r[g0], a1 = zin_r[g0 + 4096];
        float c0 = zin_i[g0], c1 = zin_i[g0 + 4096];
        *(unsigned*)&sZr[r][xx][ci] = pack_bf16x2(a0, a1);
        *(unsigned*)&sZi[r][xx][ci] = pack_bf16x2(c0, c1);
    }
    __syncthreads();

    const int wave = tid >> 5;
    const int lane = tid & 31;
    const int ln16 = lane & 15;
    const int lq   = lane >> 4;

    float wsum_dz = 0.f, wsum_nrm = 0.f;

#pragma unroll 1
    for (int rep = 0; rep < 2; ++rep) {
        const int T = wave * 2 + rep;            // 16 tiles: 4 co x 4 px
        const int co_base = (T >> 2) * 16;
        const int px_base = (T & 3) * 16;

        v8f acc_cr = {0.f, 0.f, 0.f, 0.f, 0.f, 0.f, 0.f, 0.f};
        v8f acc_ci = {0.f, 0.f, 0.f, 0.f, 0.f, 0.f, 0.f, 0.f};

        // per-tile invariant weight lane base: element ((co*9+tap)*64 + lq*8)
        const int co = co_base + ln16;
        const size_t wlane = (size_t)co * 9 * 64 + lq * 8;

        // Keep the K loop ROLLED: bounded VGPR live set, no spills.
#pragma unroll 1
        for (int tap = 0; tap < 9; ++tap) {
            const int r  = tap / 3;
            const int dx = tap % 3 - 1;
            const int xw = (px_base + ln16 + dx + 64) & 63;
            const size_t wb0 = wlane + (size_t)tap * 64;
#pragma unroll
            for (int h = 0; h < 2; ++h) {
                const int ci0 = h * 32;
                // B fragments (K x N = 32 x 16): lane = column, K = 8*lq+s, 16+8*lq+(s-8)
                Frag16 bzr, bzi;
                bzr.h[0] = *(const v8bf*)&sZr[r][xw][ci0 + lq * 8];
                bzr.h[1] = *(const v8bf*)&sZr[r][xw][ci0 + lq * 8 + 16];
                bzi.h[0] = *(const v8bf*)&sZi[r][xw][ci0 + lq * 8];
                bzi.h[1] = *(const v8bf*)&sZi[r][xw][ci0 + lq * 8 + 16];

                // A fragments (M x K = 16 x 32) per documented bf16 layout
                const size_t wb = wb0 + ci0;
                Frag16 awr, awi, awn;
                awr.h[0] = *(const v8bf*)&wrn[wb];
                awr.h[1] = *(const v8bf*)&wrn[wb + 16];
                awi.h[0] = *(const v8bf*)&win[wb];
                awi.h[1] = *(const v8bf*)&win[wb + 16];
                awn.h[0] = *(const v8bf*)&wineg[wb];
                awn.h[1] = *(const v8bf*)&wineg[wb + 16];

                // cr += Wr*zr + (-Wi)*zi ; ci += Wi*zr + Wr*zi
                acc_cr = __builtin_amdgcn_wmma_f32_16x16x32_bf16(
                    false, awr.v, false, bzr.v, (short)0, acc_cr, false, false);
                acc_cr = __builtin_amdgcn_wmma_f32_16x16x32_bf16(
                    false, awn.v, false, bzi.v, (short)0, acc_cr, false, false);
                acc_ci = __builtin_amdgcn_wmma_f32_16x16x32_bf16(
                    false, awi.v, false, bzr.v, (short)0, acc_ci, false, false);
                acc_ci = __builtin_amdgcn_wmma_f32_16x16x32_bf16(
                    false, awr.v, false, bzi.v, (short)0, acc_ci, false, false);
            }
        }

        // Epilogue in f32: D[v][lane] -> (co = co_base + v + 8*lq, x = px_base + ln16)
        const int xx = px_base + ln16;
#pragma unroll
        for (int v = 0; v < 8; ++v) {
            const int cc = co_base + v + lq * 8;
            const size_t g = (((size_t)b * CCH + cc) * HWD + y) * HWD + xx;
            const float zr_o = zin_r[g];
            const float zi_o = zin_i[g];
            const float cr  = acc_cr[v] + injr[g];   // inj_i == 0
            const float cim = acc_ci[v];
            const float mag = sqrtf(cr * cr + cim * cim + EPSV);
            const float s   = fmaxf(mag + mod_b[cc], 0.f) / mag;
            const float nzr = (1.f - alpha) * zr_o + alpha * (s * cr);
            const float nzi = (1.f - alpha) * zi_o + alpha * (s * cim);
            zout_r[g] = nzr;
            zout_i[g] = nzi;
            const float d1 = nzr - zr_o, d2 = nzi - zi_o;
            wsum_dz  += d1 * d1 + d2 * d2;
            wsum_nrm += zr_o * zr_o + zi_o * zi_o;
        }
    }

    // wave32 reduction + one atomic per wave
#pragma unroll
    for (int m = 16; m >= 1; m >>= 1) {
        wsum_dz  += __shfl_xor(wsum_dz,  m, 32);
        wsum_nrm += __shfl_xor(wsum_nrm, m, 32);
    }
    if (lane == 0) {
        atomicAdd(dz2_slot,  wsum_dz);
        atomicAdd(nrm2_slot, wsum_nrm);
    }
}

// =====================================================================
// 5) Pooled magnitude: pooled[b*64+c] = mean_{y,x} |z|
// =====================================================================
__global__ void __launch_bounds__(256)
pool_kernel(const float* __restrict__ zr, const float* __restrict__ zi,
            float* __restrict__ pooled)
{
    const int bc = blockIdx.x;
    const size_t base = (size_t)bc * 4096;
    const int tid = threadIdx.x;
    float s = 0.f;
    for (int p = tid; p < 4096; p += 256) {
        float a = zr[base + p], c = zi[base + p];
        s += sqrtf(a * a + c * c);
    }
#pragma unroll
    for (int m = 16; m >= 1; m >>= 1) s += __shfl_xor(s, m, 32);
    __shared__ float r8[8];
    if ((tid & 31) == 0) r8[tid >> 5] = s;
    __syncthreads();
    if (tid == 0) {
        float t = 0.f;
        for (int i = 0; i < 8; ++i) t += r8[i];
        pooled[bc] = t * (1.f / 4096.f);
    }
}

// =====================================================================
// 6) Logits + history finalize
// =====================================================================
__global__ void __launch_bounds__(512)
logits_kernel(const float* __restrict__ pooled, const float* __restrict__ ro_w,
              const float* __restrict__ ro_b, const float* __restrict__ dz2,
              const float* __restrict__ nrm2, float* __restrict__ out)
{
    const int tid = threadIdx.x;
    if (tid < BATCH * NCLS) {
        int b = tid / NCLS, n = tid % NCLS;
        float acc = ro_b[n];
        for (int c = 0; c < CCH; ++c) acc += pooled[b * CCH + c] * ro_w[n * CCH + c];
        out[b * NCLS + n] = acc;
    } else if (tid < BATCH * NCLS + NSTEP) {
        int t = tid - BATCH * NCLS;
        out[BATCH * NCLS + t] = sqrtf(dz2[t]) / (sqrtf(nrm2[t]) + EPSV);
    }
}

// =====================================================================
extern "C" void kernel_launch(void* const* d_in, const int* in_sizes, int n_in,
                              void* d_out, int out_size, void* d_ws, size_t ws_size,
                              hipStream_t stream)
{
    (void)in_sizes; (void)n_in; (void)out_size; (void)ws_size;
    const float* x       = (const float*)d_in[0];
    const float* enc_w   = (const float*)d_in[1];
    const float* enc_b   = (const float*)d_in[2];
    const float* kern_wr = (const float*)d_in[3];
    const float* kern_wi = (const float*)d_in[4];
    const float* mod_b   = (const float*)d_in[5];
    const float* alpha   = (const float*)d_in[6];
    const float* ro_w    = (const float*)d_in[7];
    const float* ro_b    = (const float*)d_in[8];
    float* out = (float*)d_out;

    const size_t ZE = (size_t)BATCH * CCH * HWD * HWD;   // 8388608
    float* zr_a = (float*)d_ws;
    float* zi_a = zr_a + ZE;
    float* zr_b = zi_a + ZE;
    float* zi_b = zr_b + ZE;
    float* injr = zi_b + ZE;
    bf16_t* wmats = (bf16_t*)(injr + ZE);
    bf16_t* wrn   = wmats;
    bf16_t* win   = wmats + 36864;
    bf16_t* wineg = wmats + 2 * 36864;
    float* red    = (float*)(wmats + 3 * 36864);
    float* dz2    = red;
    float* nrm2   = red + NSTEP;
    float* sigma  = red + 2 * NSTEP;
    float* pooled = red + 2 * NSTEP + 1;

    encode_kernel<<<dim3(HWD, BATCH), 256, 0, stream>>>(x, enc_w, enc_b, injr, zr_a, zi_a);
    spectral_kernel<<<1, 576, 0, stream>>>(kern_wr, kern_wi, sigma);
    prep_kernel<<<144, 256, 0, stream>>>(kern_wr, kern_wi, sigma, wrn, win, wineg, dz2, nrm2);

    float* cur_r = zr_a; float* cur_i = zi_a;
    float* nxt_r = zr_b; float* nxt_i = zi_b;
    for (int t = 0; t < NSTEP; ++t) {
        step_kernel<<<dim3(HWD, BATCH), 256, 0, stream>>>(
            cur_r, cur_i, nxt_r, nxt_i, injr, wrn, win, wineg,
            mod_b, alpha, dz2 + t, nrm2 + t);
        float* tr = cur_r; cur_r = nxt_r; nxt_r = tr;
        float* ti = cur_i; cur_i = nxt_i; nxt_i = ti;
    }

    pool_kernel<<<BATCH * CCH, 256, 0, stream>>>(cur_r, cur_i, pooled);
    logits_kernel<<<1, 512, 0, stream>>>(pooled, ro_w, ro_b, dz2, nrm2, out);
}
